// SlotAttentionLayer_48369921687885
// MI455X (gfx1250) — matrix operand order repaired and checked by reading, not compiled, not run on hardware
//
#include <hip/hip_runtime.h>
#include <hip/hip_bf16.h>

typedef __attribute__((ext_vector_type(16))) __bf16 v16bf;
typedef __attribute__((ext_vector_type(8)))  __bf16 v8bf;
typedef __attribute__((ext_vector_type(8)))  float  v8f;

#define DEV __device__ __forceinline__

// ---------------------------------------------------------------------------
// Problem constants (from the reference)
// ---------------------------------------------------------------------------
#define CB 2
#define CL 2048
#define CS 2048
#define CD 1024
#define CH 16
#define CNS 8
#define CNSLOT 512
#define CHD 64

// ---------------------------------------------------------------------------
// fp32 -> bf16 conversion (weights)
// ---------------------------------------------------------------------------
__global__ void f32_to_bf16_kernel(const float* __restrict__ src,
                                   __bf16* __restrict__ dst, int n) {
  for (int i = blockIdx.x * blockDim.x + threadIdx.x; i < n;
       i += gridDim.x * blockDim.x)
    dst[i] = (__bf16)src[i];
}

// ---------------------------------------------------------------------------
// Row LayerNorm (optionally with SiLU(C @ Wpe^T) multiplicative gate fused),
// fp32 in -> bf16 out.  One block (256 thr) per row of D=1024.
// ---------------------------------------------------------------------------
__global__ __launch_bounds__(256)
void rowln_kernel(const float* __restrict__ X,      // [rows, D]
                  const float* __restrict__ Cg,     // [rows, NS] or nullptr
                  const float* __restrict__ Wpe,    // [D, NS]
                  const float* __restrict__ gamma,
                  const float* __restrict__ beta,
                  __bf16* __restrict__ out) {
  const int row = blockIdx.x;
  __shared__ float red[256];
  __shared__ float cs[CNS];
  if (Cg != nullptr && threadIdx.x < CNS)
    cs[threadIdx.x] = Cg[row * CNS + threadIdx.x];
  __syncthreads();

  float vals[4];
  float sum = 0.f;
#pragma unroll
  for (int i = 0; i < 4; ++i) {
    int d = i * 256 + threadIdx.x;
    float x = X[(size_t)row * CD + d];
    if (Cg != nullptr) {
      float g = 0.f;
#pragma unroll
      for (int m = 0; m < CNS; ++m) g += cs[m] * Wpe[d * CNS + m];
      g = g / (1.f + __expf(-g));       // SiLU
      x *= g;
    }
    vals[i] = x;
    sum += x;
  }
  red[threadIdx.x] = sum; __syncthreads();
  for (int s = 128; s > 0; s >>= 1) {
    if (threadIdx.x < s) red[threadIdx.x] += red[threadIdx.x + s];
    __syncthreads();
  }
  float mu = red[0] * (1.f / CD);
  __syncthreads();
  float s2 = 0.f;
#pragma unroll
  for (int i = 0; i < 4; ++i) { float d = vals[i] - mu; s2 += d * d; }
  red[threadIdx.x] = s2; __syncthreads();
  for (int s = 128; s > 0; s >>= 1) {
    if (threadIdx.x < s) red[threadIdx.x] += red[threadIdx.x + s];
    __syncthreads();
  }
  float rstd = rsqrtf(red[0] * (1.f / CD) + 1e-5f);
#pragma unroll
  for (int i = 0; i < 4; ++i) {
    int d = i * 256 + threadIdx.x;
    out[(size_t)row * CD + d] = (__bf16)((vals[i] - mu) * rstd * gamma[d] + beta[d]);
  }
}

// ---------------------------------------------------------------------------
// WMMA fragment loaders (contiguous k order, shared by A and B)
// A-frag (16x32, row-major src): elem j -> K = (lane<16?0:16)+j
// B-frag (src = W[N,K] row-major, B[k][n]=W[n][k]): elem j -> K = (lane<16?0:16)+j
// ---------------------------------------------------------------------------
DEV v16bf load_a_frag(const __bf16* __restrict__ p /*row ptr incl k0*/, int lane) {
  int off = (lane < 16) ? 0 : 16;
  v8bf lo = *(const v8bf*)(p + off);
  v8bf hi = *(const v8bf*)(p + 8 + off);
  v16bf a;
#pragma unroll
  for (int j = 0; j < 8; ++j) { a[j] = lo[j]; a[8 + j] = hi[j]; }
  return a;
}

DEV v16bf load_b_frag(const __bf16* __restrict__ p /*W row ptr incl k0+koff*/) {
  v8bf lo = *(const v8bf*)(p);
  v8bf hi = *(const v8bf*)(p + 8);
  v16bf b;
#pragma unroll
  for (int j = 0; j < 8; ++j) { b[j] = lo[j]; b[8 + j] = hi[j]; }
  return b;
}

// ---------------------------------------------------------------------------
// Generic bf16 GEMM:  C[M,N] = scale * (A[M,K] @ W[N,K]^T)
// One wave computes a 32(M) x 64(N) tile: 2 A-frags + 4 B-frags per K=32 step,
// software-pipelined (double-buffered fragments) so the next step's 6 fragment
// loads are in flight while the current step's 8 WMMAs issue.
// ---------------------------------------------------------------------------
struct Frags { v16bf a[2]; v16bf b[4]; };

DEV void load_frags(Frags& f, const __bf16* __restrict__ A, int lda,
                    const __bf16* __restrict__ W, int ldw,
                    int m0, int n0, int k0, int lane) {
  const int arow  = lane & 15;
  const int bn    = lane & 15;
  const int bkoff = (lane < 16) ? 0 : 16;
#pragma unroll
  for (int ar = 0; ar < 2; ++ar)
    f.a[ar] = load_a_frag(A + (size_t)(m0 + ar * 16 + arow) * lda + k0, lane);
#pragma unroll
  for (int t = 0; t < 4; ++t)
    f.b[t] = load_b_frag(W + (size_t)(n0 + t * 16 + bn) * ldw + k0 + bkoff);
}

DEV void wmma8(const Frags& f, v8f acc[2][4]) {
#pragma unroll
  for (int ar = 0; ar < 2; ++ar)
#pragma unroll
    for (int t = 0; t < 4; ++t)
      acc[ar][t] = __builtin_amdgcn_wmma_f32_16x16x32_bf16(
          false, f.a[ar], false, f.b[t], (short)0, acc[ar][t], false, false);
}

__global__ __launch_bounds__(256)
void gemm_bf16_kernel(const __bf16* __restrict__ A, int lda,
                      const __bf16* __restrict__ W, int ldw,
                      __bf16* __restrict__ Cb, float* __restrict__ Cf, int ldc,
                      int M, int N, int K, float out_scale) {
  int wave = (blockIdx.x * blockDim.x + threadIdx.x) >> 5;
  int lane = threadIdx.x & 31;
  int tilesN = N >> 6;
  int tm = wave / tilesN;               // 32-row tiles
  int tn = wave % tilesN;               // 64-col tiles
  if (tm * 32 >= M) return;
  const int m0 = tm * 32, n0 = tn * 64;

  const v8f zero = {0.f, 0.f, 0.f, 0.f, 0.f, 0.f, 0.f, 0.f};
  v8f acc[2][4] = {{zero, zero, zero, zero}, {zero, zero, zero, zero}};

  Frags f0, f1;
  load_frags(f0, A, lda, W, ldw, m0, n0, 0, lane);
  int k0 = 0;
  for (; k0 + 64 < K; k0 += 64) {       // K is a multiple of 64 here
    load_frags(f1, A, lda, W, ldw, m0, n0, k0 + 32, lane);
    wmma8(f0, acc);
    load_frags(f0, A, lda, W, ldw, m0, n0, k0 + 64, lane);
    wmma8(f1, acc);
  }
  load_frags(f1, A, lda, W, ldw, m0, n0, k0 + 32, lane);
  wmma8(f0, acc);
  wmma8(f1, acc);

  int crow = (lane < 16) ? 0 : 8;
#pragma unroll
  for (int ar = 0; ar < 2; ++ar) {
#pragma unroll
    for (int t = 0; t < 4; ++t) {
#pragma unroll
      for (int r = 0; r < 8; ++r) {
        size_t m = (size_t)m0 + ar * 16 + crow + r;
        size_t n = (size_t)n0 + t * 16 + (lane & 15);
        float v = acc[ar][t][r] * out_scale;
        if (Cf) Cf[m * ldc + n] = v;
        else    Cb[m * ldc + n] = (__bf16)v;
      }
    }
  }
}

// ---------------------------------------------------------------------------
// Gather slots + multiplicative EMA gate.
//   gate[s,d] = SiLU( sum_m rho_m^(S-1-s) * Wpe[d,m] )
//   Kg[b,s,d] = bf16( Ek[idx[b,s], d] * gate[s,d] )   (row-major, for QK^T)
//   Vt[b,d,s] = bf16( Ev[idx[b,s], d] * gate[s,d] )   (transposed, for P@V)
// ---------------------------------------------------------------------------
__global__ __launch_bounds__(256)
void gather_gate_kernel(const int* __restrict__ x_idx,
                        const __bf16* __restrict__ Ek,
                        const __bf16* __restrict__ Ev,
                        const float* __restrict__ rhos,
                        const float* __restrict__ Wpe,   // [D, NS]
                        __bf16* __restrict__ Kg,         // [B,S,D]
                        __bf16* __restrict__ Vt) {       // [B,D,S]
  int s = blockIdx.x % CS;
  int b = blockIdx.x / CS;
  float age = (float)(CS - 1 - s);
  float c[CNS];
#pragma unroll
  for (int m = 0; m < CNS; ++m) c[m] = __powf(rhos[m], age);
  int idx = x_idx[b * CS + s];
  for (int d = threadIdx.x; d < CD; d += blockDim.x) {
    float g = 0.f;
#pragma unroll
    for (int m = 0; m < CNS; ++m) g += c[m] * Wpe[d * CNS + m];
    g = g / (1.f + __expf(-g));
    float kv = (float)Ek[(size_t)idx * CD + d] * g;
    float vv = (float)Ev[(size_t)idx * CD + d] * g;
    Kg[((size_t)b * CS + s) * CD + d] = (__bf16)kv;
    Vt[((size_t)b * CD + d) * CS + s] = (__bf16)vv;
  }
}

// ---------------------------------------------------------------------------
// Flash attention: one wave per (b, h, 16-row L tile).  S in chunks of 32.
// scores = Q(16x64) @ K^T -> 4 WMMAs ; V fragments prefetched before the
// online softmax so their L2 latency hides under the exp/shuffle VALU work ;
// P re-layout C-frag -> A-frag via a wave-private LDS tile ; O += P @ V.
// Softmax 1/sqrt(HD) is pre-folded into Q.
// ---------------------------------------------------------------------------
__global__ __launch_bounds__(256)
void attn_kernel(const __bf16* __restrict__ Q,   // [B*L, D]
                 const __bf16* __restrict__ Kg,  // [B,S,D]
                 const __bf16* __restrict__ Vt,  // [B,D,S]
                 __bf16* __restrict__ O) {       // [B*L, D]
  __shared__ __align__(16) __bf16 lds_p[8][16 * 32];

  int lane  = threadIdx.x & 31;
  int wslot = threadIdx.x >> 5;
  int wave  = (blockIdx.x * blockDim.x + threadIdx.x) >> 5;
  const int ltiles = CL / 16;
  int lt = wave % ltiles;
  int bh = wave / ltiles;
  int h  = bh % CH;
  int b  = bh / CH;

  const int arow  = lane & 15;
  const int bn    = lane & 15;
  const int bkoff = (lane < 16) ? 0 : 16;

  // Q fragments (K-dim = HD = 64 -> two K=32 fragments)
  v16bf qa[2];
#pragma unroll
  for (int kb = 0; kb < 2; ++kb) {
    const __bf16* p = Q + (size_t)(b * CL + lt * 16 + arow) * CD + h * CHD + kb * 32;
    qa[kb] = load_a_frag(p, lane);
  }

  float mrow[8], lrow[8];
#pragma unroll
  for (int r = 0; r < 8; ++r) { mrow[r] = -3.0e38f; lrow[r] = 0.f; }
  const v8f zero = {0.f, 0.f, 0.f, 0.f, 0.f, 0.f, 0.f, 0.f};
  v8f oacc[4] = {zero, zero, zero, zero};

  __bf16* lp = &lds_p[wslot][0];

  for (int s0 = 0; s0 < CS; s0 += 32) {
    // ---- scores: two 16x16 column sub-tiles, accumulate over HD ----
    v8f sf[2] = {zero, zero};
#pragma unroll
    for (int sub = 0; sub < 2; ++sub) {
#pragma unroll
      for (int kb = 0; kb < 2; ++kb) {
        const __bf16* kp = Kg + ((size_t)b * CS + s0 + sub * 16 + bn) * CD
                              + h * CHD + kb * 32 + bkoff;
        v16bf kf = load_b_frag(kp);
        sf[sub] = __builtin_amdgcn_wmma_f32_16x16x32_bf16(
            false, qa[kb], false, kf, (short)0, sf[sub], false, false);
      }
    }
    // ---- prefetch V fragments (independent of softmax / LDS round-trip) ----
    v16bf vf[4];
#pragma unroll
    for (int t = 0; t < 4; ++t) {
      const __bf16* vp = Vt + ((size_t)b * CD + h * CHD + t * 16 + bn) * CS
                            + s0 + bkoff;
      vf[t] = load_b_frag(vp);
    }
    // ---- online softmax (row = (lane<16?0:8)+r, cols across 16-lane half) ----
    float pv0[8], pv1[8];
#pragma unroll
    for (int r = 0; r < 8; ++r) {
      float mx = fmaxf(sf[0][r], sf[1][r]);
#pragma unroll
      for (int off = 8; off >= 1; off >>= 1)
        mx = fmaxf(mx, __shfl_xor(mx, off, 32));
      float mnew  = fmaxf(mrow[r], mx);
      float alpha = __expf(mrow[r] - mnew);
      float p0 = __expf(sf[0][r] - mnew);
      float p1 = __expf(sf[1][r] - mnew);
      float ps = p0 + p1;
#pragma unroll
      for (int off = 8; off >= 1; off >>= 1)
        ps += __shfl_xor(ps, off, 32);
      lrow[r] = lrow[r] * alpha + ps;
      mrow[r] = mnew;
      pv0[r] = p0; pv1[r] = p1;
#pragma unroll
      for (int t = 0; t < 4; ++t) oacc[t][r] *= alpha;
    }
    // ---- P: C-frag layout -> row-major LDS tile -> A-frag layout ----
    {
      int prow = (lane < 16) ? 0 : 8;
#pragma unroll
      for (int r = 0; r < 8; ++r) {
        lp[(prow + r) * 32 + bn]      = (__bf16)pv0[r];
        lp[(prow + r) * 32 + 16 + bn] = (__bf16)pv1[r];
      }
    }
    asm volatile("s_wait_dscnt 0x0" ::: "memory");
    __builtin_amdgcn_wave_barrier();
    v16bf pa = load_a_frag(lp + arow * 32, lane);
    asm volatile("s_wait_dscnt 0x0" ::: "memory");
    __builtin_amdgcn_wave_barrier();
    // ---- O += P @ V (fragments already resident) ----
#pragma unroll
    for (int t = 0; t < 4; ++t) {
      oacc[t] = __builtin_amdgcn_wmma_f32_16x16x32_bf16(
          false, pa, false, vf[t], (short)0, oacc[t], false, false);
    }
  }
  // ---- normalize by row sums and store bf16 for the output projection ----
  int crow = (lane < 16) ? 0 : 8;
#pragma unroll
  for (int t = 0; t < 4; ++t) {
#pragma unroll
    for (int r = 0; r < 8; ++r) {
      float v = oacc[t][r] / lrow[r];
      O[(size_t)(b * CL + lt * 16 + crow + r) * CD + h * CHD + t * 16 + bn] = (__bf16)v;
    }
  }
}

// ---------------------------------------------------------------------------
// Host-side orchestration
// ---------------------------------------------------------------------------
static inline void launch_gemm(const __bf16* A, int lda, const __bf16* W, int ldw,
                               __bf16* Cb, float* Cf, int ldc,
                               int M, int N, int K, float scale, hipStream_t s) {
  int waves = (M / 32) * (N / 64);
  int grid  = (waves * 32 + 255) / 256;
  gemm_bf16_kernel<<<grid, 256, 0, s>>>(A, lda, W, ldw, Cb, Cf, ldc, M, N, K, scale);
}

extern "C" void kernel_launch(void* const* d_in, const int* in_sizes, int n_in,
                              void* d_out, int out_size, void* d_ws, size_t ws_size,
                              hipStream_t stream) {
  (void)in_sizes; (void)n_in; (void)out_size; (void)ws_size;
  const float* x_q     = (const float*)d_in[0];
  const int*   x_idx   = (const int*)  d_in[1];
  const float* E_slots = (const float*)d_in[2];
  const float* rhos    = (const float*)d_in[3];
  const float* C_seq   = (const float*)d_in[4];
  const float* Wq      = (const float*)d_in[5];
  const float* Wk      = (const float*)d_in[6];
  const float* Wv      = (const float*)d_in[7];
  const float* Wo      = (const float*)d_in[8];
  const float* Wpe     = (const float*)d_in[9];
  const float* ln_kv_g = (const float*)d_in[10];
  const float* ln_kv_b = (const float*)d_in[11];
  const float* ln_q_g  = (const float*)d_in[12];
  const float* ln_q_b  = (const float*)d_in[13];
  float* out = (float*)d_out;

  // workspace carve-up (all 256B aligned)
  char* w = (char*)d_ws;
  auto carve = [&](size_t bytes) -> char* {
    char* p = w; w += (bytes + 255) & ~(size_t)255; return p;
  };
  __bf16* memb = (__bf16*)carve((size_t)CNSLOT * CD * 2);     // LN(E_slots)
  __bf16* Wqb  = (__bf16*)carve((size_t)CD * CD * 2);
  __bf16* Wkb  = (__bf16*)carve((size_t)CD * CD * 2);
  __bf16* Wvb  = (__bf16*)carve((size_t)CD * CD * 2);
  __bf16* Wob  = (__bf16*)carve((size_t)CD * CD * 2);
  __bf16* Ek   = (__bf16*)carve((size_t)CNSLOT * CD * 2);
  __bf16* Ev   = (__bf16*)carve((size_t)CNSLOT * CD * 2);
  __bf16* Kg   = (__bf16*)carve((size_t)CB * CS * CD * 2);    // gated K [B,S,D]
  __bf16* Vt   = (__bf16*)carve((size_t)CB * CD * CS * 2);    // gated V^T [B,D,S]
  __bf16* xln  = (__bf16*)carve((size_t)CB * CL * CD * 2);    // LN(x_q*gate_q)
  __bf16* Qs   = (__bf16*)carve((size_t)CB * CL * CD * 2);    // scaled Q
  __bf16* AttO = (__bf16*)carve((size_t)CB * CL * CD * 2);    // attention output

  // 1. mem = LN(E_slots)  -> bf16
  rowln_kernel<<<CNSLOT, 256, 0, stream>>>(E_slots, nullptr, nullptr,
                                           ln_kv_g, ln_kv_b, memb);
  // 2. weights -> bf16
  f32_to_bf16_kernel<<<2048, 256, 0, stream>>>(Wq, Wqb, CD * CD);
  f32_to_bf16_kernel<<<2048, 256, 0, stream>>>(Wk, Wkb, CD * CD);
  f32_to_bf16_kernel<<<2048, 256, 0, stream>>>(Wv, Wvb, CD * CD);
  f32_to_bf16_kernel<<<2048, 256, 0, stream>>>(Wo, Wob, CD * CD);
  // 3. Ek = mem @ Wk^T ; Ev = mem @ Wv^T   (512x1024x1024, WMMA bf16)
  launch_gemm(memb, CD, Wkb, CD, Ek, nullptr, CD, CNSLOT, CD, CD, 1.f, stream);
  launch_gemm(memb, CD, Wvb, CD, Ev, nullptr, CD, CNSLOT, CD, CD, 1.f, stream);
  // 4. gather + EMA gate -> Kg (row-major), Vt (transposed)
  gather_gate_kernel<<<CB * CS, 256, 0, stream>>>(x_idx, Ek, Ev, rhos, Wpe, Kg, Vt);
  // 5. xln = LN(x_q * SiLU(C_seq @ Wpe^T))
  rowln_kernel<<<CB * CL, 256, 0, stream>>>(x_q, C_seq, Wpe, ln_q_g, ln_q_b, xln);
  // 6. Q = (xln @ Wq^T) * (1/sqrt(HD))   (4096x1024x1024, WMMA bf16)
  launch_gemm(xln, CD, Wqb, CD, Qs, nullptr, CD, CB * CL, CD, CD, 0.125f, stream);
  // 7. flash attention (WMMA bf16, online softmax)
  {
    int waves = CB * CH * (CL / 16);           // 4096 waves
    int grid  = (waves * 32 + 255) / 256;      // 512 blocks
    attn_kernel<<<grid, 256, 0, stream>>>(Qs, Kg, Vt, AttO);
  }
  // 8. out = AttO @ Wo^T  -> fp32 d_out
  launch_gemm(AttO, CD, Wob, CD, nullptr, out, CD, CB * CL, CD, CD, 1.f, stream);
}
